// VibrotactileSensor_44092134260937
// MI455X (gfx1250) — matrix operand order repaired and checked by reading, not compiled, not run on hardware
//
#include <hip/hip_runtime.h>
#include <math.h>

typedef __attribute__((ext_vector_type(16))) _Float16 v16h;
typedef __attribute__((ext_vector_type(8)))  _Float16 v8h;
typedef __attribute__((ext_vector_type(8)))  float    v8f;
typedef __attribute__((ext_vector_type(4)))  float    v4f;
typedef __attribute__((ext_vector_type(2)))  float    v2f;

#define T_LEN 50000
#define NB 5
#define FLEN 51
#define NROWS 512
#define BLOCKS_PER_ROW 2
#define WAVES_PER_BLOCK 8
#define WAVES_PER_ROW (BLOCKS_PER_ROW * WAVES_PER_BLOCK)      // 16
// super-tile = 48 outputs (j=0..2 time-shifted filter copies in B columns n=j*5+f)
#define NSUP ((T_LEN + 47) / 48)                              // 1042
#define SUP_PER_WAVE ((NSUP + WAVES_PER_ROW - 1) / WAVES_PER_ROW) // 66
// group issued at body u reaches index <= 48u+229; statically in-bounds for u <= 1036
#define SUP_FAST_LIM 1037

// ---- checked load: clamp index, multiply by in-range mask (no branches) ----
__device__ __forceinline__ v8h load_run_checked(const float* __restrict__ xrow, int base) {
  v8h r;
#pragma unroll
  for (int j = 0; j < 8; ++j) {
    int i  = base + j;
    int ic = i < 0 ? 0 : (i > T_LEN - 1 ? T_LEN - 1 : i);
    float m = ((unsigned)i < (unsigned)T_LEN) ? 1.0f : 0.0f;
    r[j] = (_Float16)(xrow[ic] * m);
  }
  return r;
}

// ---- interior refill: 3 runs (6 DWORD-aligned b128) from ONE base address ----
// runs at +0, +64, +128 bytes; each run = 2 x b128 (+0,+16)
__device__ __forceinline__ void issue_group6(const float* __restrict__ addr,
                                             v4f& a0, v4f& a1, v4f& a2,
                                             v4f& a3, v4f& a4, v4f& a5) {
  asm volatile("global_load_b128 %0, %6, off\n\t"
               "global_load_b128 %1, %6, off offset:16\n\t"
               "global_load_b128 %2, %6, off offset:64\n\t"
               "global_load_b128 %3, %6, off offset:80\n\t"
               "global_load_b128 %4, %6, off offset:128\n\t"
               "global_load_b128 %5, %6, off offset:144"
               : "=v"(a0), "=v"(a1), "=v"(a2), "=v"(a3), "=v"(a4), "=v"(a5)
               : "v"(addr)
               : "memory");
}
// loads retire in order: <=6 outstanding means the older group of 6 is complete
__device__ __forceinline__ void wait_group(v4f& p0, v4f& p1, v4f& p2, v4f& p3, v4f& p4, v4f& p5) {
  asm volatile("s_wait_loadcnt 0x6"
               : "+v"(p0), "+v"(p1), "+v"(p2), "+v"(p3), "+v"(p4), "+v"(p5)::"memory");
}
__device__ __forceinline__ void drain_group(v4f& p0, v4f& p1, v4f& p2, v4f& p3, v4f& p4, v4f& p5) {
  asm volatile("s_wait_loadcnt 0x0"
               : "+v"(p0), "+v"(p1), "+v"(p2), "+v"(p3), "+v"(p4), "+v"(p5)::"memory");
}
__device__ __forceinline__ v8h cvt_run(v4f a, v4f b) {
  v8h r;
#pragma unroll
  for (int j = 0; j < 4; ++j) {
    r[j]     = (_Float16)a[j];
    r[j + 4] = (_Float16)b[j];
  }
  return r;
}

__device__ __forceinline__ v16h cat8(v8h lo, v8h hi) {
  v16h a;
#pragma unroll
  for (int j = 0; j < 8; ++j) { a[j] = lo[j]; a[j + 8] = hi[j]; }
  return a;
}

// stats: v_pk_add_f32 (sum), v_pk_fma_f32 (sumsq), adds with |mod| (abs), v_max3 (max)
#define ACC_STATS(C)                                                     \
  {                                                                      \
    _Pragma("unroll")                                                    \
    for (int q = 0; q < 4; ++q) {                                        \
      v2f d2;                                                            \
      d2.x = (C)[2 * q];                                                 \
      d2.y = (C)[2 * q + 1];                                             \
      acc_sum += d2;                                                     \
      acc_sq  += d2 * d2;                                                \
      float a0 = fabsf(d2.x), a1 = fabsf(d2.y);                          \
      ab0 += a0;                                                         \
      ab1 += a1;                                                         \
      s_max = fmaxf(s_max, fmaxf(a0, a1));                               \
    }                                                                    \
  }

#define ACC_STATS_MASKED(C, MK)                                          \
  {                                                                      \
    _Pragma("unroll")                                                    \
    for (int q = 0; q < 4; ++q) {                                        \
      v2f d2;                                                            \
      d2.x = (C)[2 * q] * (MK);                                          \
      d2.y = (C)[2 * q + 1] * (MK);                                      \
      acc_sum += d2;                                                     \
      acc_sq  += d2 * d2;                                                \
      float a0 = fabsf(d2.x), a1 = fabsf(d2.y);                          \
      ab0 += a0;                                                         \
      ab1 += a1;                                                         \
      s_max = fmaxf(s_max, fmaxf(a0, a1));                               \
    }                                                                    \
  }

// one super-tile: 3 chained WMMAs = 48 outputs, one stats pass, retire 3 runs, issue 3
#define SUP_BODY(A0, A1, A2, A3, A4, A5, P0, P1, P2, P3, P4, P5)               \
  {                                                                            \
    v8f c = {};                                                                \
    c = __builtin_amdgcn_wmma_f32_16x16x32_f16(false, cat8(A0, A1), false, B0, \
                                               (short)0, c, false, false);     \
    c = __builtin_amdgcn_wmma_f32_16x16x32_f16(false, cat8(A2, A3), false, B1, \
                                               (short)0, c, false, false);     \
    c = __builtin_amdgcn_wmma_f32_16x16x32_f16(false, cat8(A4, A5), false, B2, \
                                               (short)0, c, false, false);     \
    ACC_STATS(c);                                                              \
    wait_group(P0, P1, P2, P3, P4, P5);                                        \
    A0 = cvt_run(P0, P1);                                                      \
    A1 = cvt_run(P2, P3);                                                      \
    A2 = cvt_run(P4, P5);                                                      \
    issue_group6(ap, P0, P1, P2, P3, P4, P5);                                  \
    ap += 48;                                                                  \
  }

__global__ __launch_bounds__(256) void vib_conv_stats(const float* __restrict__ x,
                                                      float* __restrict__ partial) {
  __shared__ float h_pad[NB][64];                       // filter taps, zero padded
  __shared__ float blkpart[WAVES_PER_BLOCK][NB][4];

  const int tid = threadIdx.x;

  // ---- build filters (fp64, numpy recipe) ----
  for (int i = tid; i < NB * 64; i += 256) (&h_pad[0][0])[i] = 0.0f;
  __syncthreads();
  if (tid < NB) {
    const double PI = 3.14159265358979323846;
    const double blo[NB] = {0.0, 50.0, 200.0, 500.0, 1000.0};
    const double bhi[NB] = {50.0, 200.0, 500.0, 1000.0, 2000.0};
    const double nyq = 10000.0 / 2.0;
    double h[FLEN];
    for (int k = 0; k < FLEN; ++k) h[k] = 0.0;
    double fr[2] = {blo[tid] / nyq, bhi[tid] / nyq};
    for (int i = 0; i < 2; ++i) {
      double f = fr[i];
      if (f > 0.0) {
        double sgn = (i == 0) ? 1.0 : -1.0;
        for (int k = 0; k < FLEN; ++k) {
          double d = (double)(k - FLEN / 2);
          double s = (k == FLEN / 2) ? 2.0 * f : sin(2.0 * PI * f * d) / (PI * d);
          h[k] += sgn * s;
        }
      }
    }
    double sum = 0.0;
    for (int k = 0; k < FLEN; ++k) {
      double w = 0.54 - 0.46 * cos(2.0 * PI * (double)k / (double)(FLEN - 1));
      h[k] *= w;
      sum += h[k];
    }
    for (int k = 0; k < FLEN; ++k) h_pad[tid][k] = (float)(h[k] / sum);
  }
  __syncthreads();

  const int lane = tid & 31;
  const int wave = tid >> 5;

  // ---- B fragments: column n = j*5+f holds filter f shifted by 16j; K = 0..95 ----
  const int n  = lane & 15;
  const int kb = (lane < 16) ? 0 : 16;
  const int f_lane = n % NB;
  const int j_lane = n / NB;          // 0..2 (3 => dead column 15)
  v16h B0, B1, B2;
#pragma unroll
  for (int j = 0; j < 16; ++j) {
#pragma unroll
    for (int cchunk = 0; cchunk < 3; ++cchunk) {
      int k  = 32 * cchunk + (kb >> 1) + (j & 7) + ((j & 8) << 1);
      int kk = k - 16 * j_lane;
      float hv = 0.0f;
      if (n < 15 && kk >= 0 && kk < FLEN) hv = h_pad[f_lane][kk];
      if (cchunk == 0) B0[j] = (_Float16)hv;
      if (cchunk == 1) B1[j] = (_Float16)hv;
      if (cchunk == 2) B2[j] = (_Float16)hv;
    }
  }

  // ---- work partition ----
  const int row      = blockIdx.x / BLOCKS_PER_ROW;
  const int blkInRow = blockIdx.x % BLOCKS_PER_ROW;
  const int w        = blkInRow * WAVES_PER_BLOCK + wave;
  const int u_beg    = w * SUP_PER_WAVE;
  int u_end          = u_beg + SUP_PER_WAVE;
  if (u_end > NSUP) u_end = NSUP;
  int u_fast         = u_end < SUP_FAST_LIM ? u_end : SUP_FAST_LIM;

  const float* xrow = x + (size_t)row * T_LEN;
  // Hankel A-fragment: lane L<16 row M=L (K runs {0..7},{16..23} per chunk);
  // lane L>=16 row M=L-16 (K runs {8..15},{24..31}).  6 runs of 8 per super-tile.
  const int laneoff = (lane & 15) + ((lane >> 4) << 3);

  v2f acc_sum = {0.0f, 0.0f}, acc_sq = {0.0f, 0.0f};
  float ab0 = 0.0f, ab1 = 0.0f, s_max = 0.0f;

  {
    const int s0 = u_beg * 48 - (FLEN / 2);   // t0 - 25
    v8h r0 = load_run_checked(xrow, s0 + laneoff);
    v8h r1 = load_run_checked(xrow, s0 + laneoff + 16);
    v8h r2 = load_run_checked(xrow, s0 + laneoff + 32);
    v8h r3 = load_run_checked(xrow, s0 + laneoff + 48);
    v8h r4 = load_run_checked(xrow, s0 + laneoff + 64);
    v8h r5 = load_run_checked(xrow, s0 + laneoff + 80);

    // depth-2 pipelined interior super-tiles: [u_beg, u_fast)
    {
      const int Nfast = u_fast - u_beg;                 // >= 47 for every wave
      v4f pa0, pa1, pa2, pa3, pa4, pa5;
      v4f pb0, pb1, pb2, pb3, pb4, pb5;
      const float* ap = xrow + (s0 + laneoff + 96);     // first refill group
      issue_group6(ap, pa0, pa1, pa2, pa3, pa4, pa5);
      issue_group6(ap + 48, pb0, pb1, pb2, pb3, pb4, pb5);
      ap += 96;

      int j = 0;
      for (; j + 2 <= Nfast; j += 2) {                  // rotation fully renamed
        SUP_BODY(r0, r1, r2, r3, r4, r5, pa0, pa1, pa2, pa3, pa4, pa5)
        SUP_BODY(r3, r4, r5, r0, r1, r2, pb0, pb1, pb2, pb3, pb4, pb5)
      }
      if (j < Nfast) { SUP_BODY(r0, r1, r2, r3, r4, r5, pa0, pa1, pa2, pa3, pa4, pa5) }
      drain_group(pa0, pa1, pa2, pa3, pa4, pa5);        // keep pending regs safe
      drain_group(pb0, pb1, pb2, pb3, pb4, pb5);
    }

    // boundary tail super-tiles (row end) -- at most 5, last wave only
    for (int u = u_fast; u < u_end; ++u) {
      int s = u * 48 - (FLEN / 2);
      v8h q0 = load_run_checked(xrow, s + laneoff);
      v8h q1 = load_run_checked(xrow, s + laneoff + 16);
      v8h q2 = load_run_checked(xrow, s + laneoff + 32);
      v8h q3 = load_run_checked(xrow, s + laneoff + 48);
      v8h q4 = load_run_checked(xrow, s + laneoff + 64);
      v8h q5 = load_run_checked(xrow, s + laneoff + 80);
      v8f c = {};
      c = __builtin_amdgcn_wmma_f32_16x16x32_f16(false, cat8(q0, q1), false, B0,
                                                 (short)0, c, false, false);
      c = __builtin_amdgcn_wmma_f32_16x16x32_f16(false, cat8(q2, q3), false, B1,
                                                 (short)0, c, false, false);
      c = __builtin_amdgcn_wmma_f32_16x16x32_f16(false, cat8(q4, q5), false, B2,
                                                 (short)0, c, false, false);
      // T_LEN % 16 == 0 => validity is uniform per lane (per j-block)
      float mk = (u * 48 + 16 * j_lane + 15 < T_LEN) ? 1.0f : 0.0f;
      ACC_STATS_MASKED(c, mk)
    }
  }

  float s_sum = acc_sum.x + acc_sum.y;
  float s_abs = ab0 + ab1;
  float s_sq  = acc_sq.x + acc_sq.y;

  // ---- deterministic reduction ----
  // 1) rows 8..15 -> rows 0..7 (other half-wave)
  s_sum += __shfl_down(s_sum, 16, 32);
  s_abs += __shfl_down(s_abs, 16, 32);
  s_sq  += __shfl_down(s_sq, 16, 32);
  s_max  = fmaxf(s_max, __shfl_down(s_max, 16, 32));
  // 2) combine the j-copies of each band: lanes f, f+5, f+10
  {
    float a5  = __shfl_down(s_sum, 5, 32), a10 = __shfl_down(s_sum, 10, 32);
    float b5  = __shfl_down(s_abs, 5, 32), b10 = __shfl_down(s_abs, 10, 32);
    float c5  = __shfl_down(s_sq, 5, 32),  c10 = __shfl_down(s_sq, 10, 32);
    float m5  = __shfl_down(s_max, 5, 32), m10 = __shfl_down(s_max, 10, 32);
    s_sum += a5 + a10;
    s_abs += b5 + b10;
    s_sq  += c5 + c10;
    s_max  = fmaxf(s_max, fmaxf(m5, m10));
  }

  if (lane < NB) {
    blkpart[wave][lane][0] = s_sum;
    blkpart[wave][lane][1] = s_abs;
    blkpart[wave][lane][2] = s_sq;
    blkpart[wave][lane][3] = s_max;
  }
  __syncthreads();

  if (tid < NB * 4) {
    int f = tid >> 2, st = tid & 3;
    float v = blkpart[0][f][st];
    for (int q = 1; q < WAVES_PER_BLOCK; ++q) {
      float u = blkpart[q][f][st];
      v = (st == 3) ? fmaxf(v, u) : (v + u);
    }
    partial[((row * BLOCKS_PER_ROW + blkInRow) * NB + f) * 4 + st] = v;
  }
}

__global__ void vib_finalize(const float* __restrict__ partial, float* __restrict__ out) {
  int i = blockIdx.x * blockDim.x + threadIdx.x;   // over 512*NB
  if (i >= NROWS * NB) return;
  int b = i / NB, f = i % NB;
  float sum = 0.0f, sabs = 0.0f, ssq = 0.0f, mx = 0.0f;
  for (int q = 0; q < BLOCKS_PER_ROW; ++q) {
    const float* p = partial + ((b * BLOCKS_PER_ROW + q) * NB + f) * 4;
    sum += p[0]; sabs += p[1]; ssq += p[2]; mx = fmaxf(mx, p[3]);
  }
  const float Tf = (float)T_LEN;
  float mean_abs = sabs / Tf;
  float mu  = sum / Tf;
  float var = (ssq - sum * mu) / (Tf - 1.0f);
  var = fmaxf(var, 0.0f);
  float stdv = sqrtf(var);
  float rms  = sqrtf(ssq / Tf);
  float* o = out + (size_t)b * (NB * 4) + f * 4;
  o[0] = mean_abs; o[1] = stdv; o[2] = mx; o[3] = rms;
}

extern "C" void kernel_launch(void* const* d_in, const int* in_sizes, int n_in,
                              void* d_out, int out_size, void* d_ws, size_t ws_size,
                              hipStream_t stream) {
  const float* x = (const float*)d_in[0];
  float* out     = (float*)d_out;
  float* partial = (float*)d_ws;   // NROWS * BLOCKS_PER_ROW * NB * 4 floats = 80 KB

  dim3 grid(NROWS * BLOCKS_PER_ROW);
  vib_conv_stats<<<grid, 256, 0, stream>>>(x, partial);

  int nfin = NROWS * NB;
  vib_finalize<<<(nfin + 255) / 256, 256, 0, stream>>>(partial, out);
}
